// Attention_53626961658241
// MI455X (gfx1250) — compile-verified, matrix operand order checked by reading, not run on hardware
//
#include <hip/hip_runtime.h>
#include <hip/hip_bf16.h>
#include <math.h>

// ---------------------------------------------------------------------------
// Shapes (compile-time constants from the reference)
// ---------------------------------------------------------------------------
#define B_    2
#define TQ_   2048
#define TKV_  2048
#define D_    2048
#define HQ_   16
#define HKV_  4
#define HD_   128
#define GQA_  (HQ_ / HKV_)   // 4

typedef __attribute__((ext_vector_type(16))) __bf16 v16bf;
typedef __attribute__((ext_vector_type(8)))  float  v8f;
typedef __attribute__((ext_vector_type(4)))  unsigned v4u;
typedef __attribute__((ext_vector_type(8)))  unsigned v8u;

#define L2E 1.4426950408889634f
#define LOG2_10000 13.287712379549449f

// ---------------------------------------------------------------------------
// WMMA helpers (CDNA5 gfx1250: 16x16x32 bf16, f32 accumulate)
// ---------------------------------------------------------------------------
__device__ __forceinline__ v8f wmma_bf16(v16bf a, v16bf b, v8f c) {
    return __builtin_amdgcn_wmma_f32_16x16x32_bf16(
        false, a, false, b, (short)0, c, false, false);
}

// A-matrix 16x32 bf16 fragment load from LDS (ISA 7.12.2 layout).
// k is contiguous within a row -> lowers to 2x ds_load_b128.
__device__ __forceinline__ v16bf load_a_frag(const __bf16* base, int ld) {
    v16bf a;
    const int lane = threadIdx.x & 31;
    const int m  = lane & 15;
    const int hi = lane >> 4;
    const __bf16* row = base + m * ld;
#pragma unroll
    for (int r = 0; r < 8; ++r) {
        int k = (r < 4) ? (hi * 8 + 2 * r) : (16 + hi * 8 + 2 * (r - 4));
        a[2 * r]     = row[k];
        a[2 * r + 1] = row[k + 1];
    }
    return a;
}

// B-matrix 32x16 bf16 fragment; element [k][n] at base[k*sk + n*sn].
// All call sites now use sk==1 (k contiguous) -> lowers to 2x ds_load_b128.
__device__ __forceinline__ v16bf load_b_frag(const __bf16* base, int sk, int sn) {
    v16bf b;
    const int lane = threadIdx.x & 31;
    const int n  = lane & 15;
    const int hi = lane >> 4;
#pragma unroll
    for (int r = 0; r < 8; ++r) {
        int k = (r < 4) ? (hi * 8 + 2 * r) : (16 + hi * 8 + 2 * (r - 4));
        b[2 * r]     = base[k * sk + n * sn];
        b[2 * r + 1] = base[(k + 1) * sk + n * sn];
    }
    return b;
}

// ---------------------------------------------------------------------------
// Tensor Data Mover: 2-D tiled load of bf16 data into LDS with in-flight row
// padding. tile_d0 = elements per row, tile_d1 = rows, stride = global row
// pitch in elements. pad codes:
//   pad_int 3 -> every 16 DWORDs (32 bf16 row)  -> LDS stride 40  (amt 3)
//   pad_int 4 -> every 32 DWORDs (64 bf16 row)  -> LDS stride 72  (amt 3)
//   pad_int 5 -> every 64 DWORDs (128 bf16 row) -> LDS stride 136 (amt 3)
// ---------------------------------------------------------------------------
__device__ __forceinline__ void tdm_load_2d(unsigned lds_off, const void* gptr,
                                            unsigned tile_d0, unsigned tile_d1,
                                            unsigned long long stride_elems,
                                            unsigned pad_int, unsigned pad_amt) {
    unsigned long long ga = (unsigned long long)(size_t)gptr;
    v4u g0;
    g0[0] = 1u;                                      // count=1, user desc
    g0[1] = lds_off;                                 // lds_addr (bytes)
    g0[2] = (unsigned)(ga & 0xFFFFFFFFull);          // global_addr[31:0]
    g0[3] = (unsigned)((ga >> 32) & 0x01FFFFFFull)   // global_addr[56:32]
          | (2u << 30);                              // type = 2 ("image")
    v8u g1;
    g1[0] = (1u << 16)            // data_size = 1 -> 2 bytes
          | (1u << 20)            // pad_enable
          | (pad_int << 22)
          | (pad_amt << 25);
    g1[1] = 0u;
    g1[2] = 0x4000u;              // tensor_dim0 = 1<<30
    g1[3] = 0x4000u               // tensor_dim1 = 1<<30
          | (tile_d0 << 16);
    g1[4] = tile_d1;              // tile_dim1 ; tile_dim2 = 0
    g1[5] = (unsigned)stride_elems;
    g1[6] = (unsigned)(stride_elems >> 32) & 0xFFFFu;
    g1[7] = 0u;
    asm volatile("tensor_load_to_lds %0, %1" :: "s"(g0), "s"(g1) : "memory");
}

// ---------------------------------------------------------------------------
// f32 -> bf16 conversion (for A operands, layout preserved)
// ---------------------------------------------------------------------------
__global__ __launch_bounds__(256)
void cvt_f32_bf16(const float* __restrict__ x, __bf16* __restrict__ y, int n) {
    int i = (blockIdx.x * 256 + threadIdx.x) * 4;
    if (i + 3 < n) {
        float4 v = *(const float4*)(x + i);
        y[i]     = (__bf16)v.x;
        y[i + 1] = (__bf16)v.y;
        y[i + 2] = (__bf16)v.z;
        y[i + 3] = (__bf16)v.w;
    }
}

// ---------------------------------------------------------------------------
// Tiled transpose + convert: out[C x R] bf16 = (in[R x C] f32)^T
// block (32,8); grid (C/32, R/32)
// ---------------------------------------------------------------------------
__global__ __launch_bounds__(256)
void transpose_cvt(const float* __restrict__ in, __bf16* __restrict__ out,
                   int R, int C) {
    __shared__ float t[32][33];
    const int tx = threadIdx.x;
    const int ty = threadIdx.y;
    const int r0 = blockIdx.y * 32;
    const int c0 = blockIdx.x * 32;
#pragma unroll
    for (int i = 0; i < 4; ++i)
        t[ty + 8 * i][tx] = in[(size_t)(r0 + ty + 8 * i) * C + c0 + tx];
    __syncthreads();
#pragma unroll
    for (int i = 0; i < 4; ++i)
        out[(size_t)(c0 + ty + 8 * i) * R + r0 + tx] = (__bf16)t[tx][ty + 8 * i];
}

// ---------------------------------------------------------------------------
// V transpose: vf (B,T,HKV,HD) f32 -> VT (B,HKV,HD,T) bf16
// block (32,8); grid (T/32, HD/32, B*HKV)
// ---------------------------------------------------------------------------
__global__ __launch_bounds__(256)
void transpose_v(const float* __restrict__ vf, __bf16* __restrict__ VT) {
    __shared__ float t[32][33];
    const int tx = threadIdx.x;
    const int ty = threadIdx.y;
    const int t0 = blockIdx.x * 32;
    const int d0 = blockIdx.y * 32;
    const int bh = blockIdx.z;
    const int b  = bh / HKV_;
    const int hk = bh % HKV_;
    const float* src = vf + (size_t)b * TQ_ * (HKV_ * HD_) + (size_t)hk * HD_;
#pragma unroll
    for (int i = 0; i < 4; ++i)
        t[ty + 8 * i][tx] =
            src[(size_t)(t0 + ty + 8 * i) * (HKV_ * HD_) + d0 + tx];
    __syncthreads();
    __bf16* dst = VT + (size_t)bh * HD_ * TKV_;
#pragma unroll
    for (int i = 0; i < 4; ++i)
        dst[(size_t)(d0 + ty + 8 * i) * TKV_ + t0 + tx] = (__bf16)t[tx][ty + 8 * i];
}

// ---------------------------------------------------------------------------
// All-bf16 NT GEMM with TDM double-buffered staging:
//   C[M,N] = X[M,K] * WT[N,K]^T     X, WT: bf16 global; C: f32
// Block 256 = 8 waves; tile 64(M) x 64(N), BK = 64 per TDM stage.
// Both LDS tiles are k-contiguous -> all fragment loads are ds_load_b128.
// ---------------------------------------------------------------------------
__global__ __launch_bounds__(256)
void gemm_bf16(const __bf16* __restrict__ X, const __bf16* __restrict__ WT,
               float* __restrict__ C, int M, int N, int K) {
    __shared__ __bf16 Xs[2][64][72];   // [m][k], TDM pad_int 4, amt 3
    __shared__ __bf16 Wn[2][64][72];   // [n][k]

    const int tid  = threadIdx.x;
    const int wv   = tid >> 5;
    const int wm   = wv & 3;
    const int wn   = wv >> 2;
    const int lane = tid & 31;
    const int ln   = lane & 15;
    const int hi   = lane >> 4;

    const int rowBase = blockIdx.y * 64;
    const int colBase = blockIdx.x * 64;

    v8f acc0 = {}; v8f acc1 = {};

    const int nStage = K >> 6;   // K / 64

    if (wv == 0) {
        tdm_load_2d((unsigned)(size_t)(void*)&Xs[0][0][0],
                    X + (size_t)rowBase * K, 64, 64, (unsigned)K, 4, 3);
        tdm_load_2d((unsigned)(size_t)(void*)&Wn[0][0][0],
                    WT + (size_t)colBase * K, 64, 64, (unsigned)K, 4, 3);
    }

    for (int s = 0; s < nStage; ++s) {
        const int p = s & 1;
        if (wv == 0) __builtin_amdgcn_s_wait_tensorcnt(0);
        __syncthreads();   // publish stage s; fences stage s-1 compute

        if (wv == 0 && s + 1 < nStage) {
            const int k1 = (s + 1) << 6;
            tdm_load_2d((unsigned)(size_t)(void*)&Xs[p ^ 1][0][0],
                        X + (size_t)rowBase * K + k1, 64, 64, (unsigned)K, 4, 3);
            tdm_load_2d((unsigned)(size_t)(void*)&Wn[p ^ 1][0][0],
                        WT + (size_t)colBase * K + k1, 64, 64, (unsigned)K, 4, 3);
        }

#pragma unroll
        for (int kc = 0; kc < 2; ++kc) {
            v16bf a  = load_a_frag(&Xs[p][wm * 16][kc * 32], 72);
            v16bf b0 = load_b_frag(&Wn[p][wn * 32][kc * 32],      1, 72);
            v16bf b1 = load_b_frag(&Wn[p][wn * 32 + 16][kc * 32], 1, 72);
            acc0 = wmma_bf16(a, b0, acc0);
            acc1 = wmma_bf16(a, b1, acc1);
        }
    }

#pragma unroll
    for (int r = 0; r < 8; ++r) {
        int gr  = rowBase + wm * 16 + r + 8 * hi;
        int gc0 = colBase + wn * 32 + ln;
        C[(size_t)gr * N + gc0]      = acc0[r];
        C[(size_t)gr * N + gc0 + 16] = acc1[r];
    }
}

// ---------------------------------------------------------------------------
// RoPE + transpose + bf16 convert:  (B,T,H,HD) f32  ->  (B,H,T,HD) bf16
// ---------------------------------------------------------------------------
__global__ __launch_bounds__(64)
void rope_transpose(const float* __restrict__ in, __bf16* __restrict__ out,
                    const int* __restrict__ pos, int T, int H) {
    const int row = blockIdx.x;
    const int h = row % H;
    const int t = (row / H) % T;
    const int b = row / (H * T);
    const int d = threadIdx.x;

    const float* src = in + ((size_t)(b * T + t) * H + h) * HD_;
    __bf16* dst = out + ((size_t)(b * H + h) * T + t) * HD_;

    float x1 = src[d];
    float x2 = src[d + 64];
    float frac = (2.0f * (float)d) / (float)HD_;
    float inv_freq = exp2f(-frac * LOG2_10000);
    float fr = (float)pos[b * T + t] * inv_freq;
    float s = sinf(fr), c = cosf(fr);
    dst[d]      = (__bf16)(x1 * c - x2 * s);
    dst[d + 64] = (__bf16)(x1 * s + x2 * c);
}

// ---------------------------------------------------------------------------
// Causal flash attention, all-WMMA, TDM double-buffered K/V staging.
//   Q: (B, HQ, TQ, HD) bf16   K: (B, HKV, TKV, HD) bf16
//   VT: (B, HKV, HD, TKV) bf16 (pre-transposed so P*V B-frags are contiguous)
//   ctx out: (B, TQ, HQ, HD) bf16
// Grid: (TQ/64, B*HQ). Block 256 = 8 waves.
// ---------------------------------------------------------------------------
__global__ __launch_bounds__(256)
void flash_attn(const __bf16* __restrict__ Q, const __bf16* __restrict__ K,
                const __bf16* __restrict__ VT, __bf16* __restrict__ ctx) {
    __shared__ __bf16 Qs[64][136];
    __shared__ __bf16 Ks[2][32][136];   // [kv][d]  (TDM pad_int 5)
    __shared__ __bf16 Vt[2][128][40];   // [d][kv]  (TDM pad_int 3)
    __shared__ __bf16 Ps[8][16 * 40];

    const int tid  = threadIdx.x;
    const int wv   = tid >> 5;
    const int wm   = wv & 3;
    const int wn   = wv >> 2;
    const int lane = tid & 31;
    const int ln   = lane & 15;
    const int hi   = lane >> 4;

    const int q0 = blockIdx.x * 64;
    const int bh = blockIdx.y;
    const int b  = bh / HQ_;
    const int h  = bh % HQ_;
    const int hk = h / GQA_;

    const __bf16* Qb = Q  + ((size_t)(b * HQ_ + h) * TQ_ + q0) * HD_;
    const __bf16* Kb = K  + ((size_t)(b * HKV_ + hk) * TKV_) * HD_;
    const __bf16* Vb = VT + ((size_t)(b * HKV_ + hk) * HD_) * TKV_;

    // Load Q tile 64x128: each thread reads 64 contiguous bytes -> b128 loads
    {
        const int qr = tid >> 2;
        const int qc = (tid & 3) * 32;
#pragma unroll
        for (int c = 0; c < 32; ++c)
            Qs[qr][qc + c] = Qb[(size_t)qr * HD_ + qc + c];
    }

    const float scale = 0.08838834764831845f;   // 1/sqrt(128)

    v8f acc[4];
#pragma unroll
    for (int c = 0; c < 4; ++c) acc[c] = (v8f){};
    float m_st[8], l_st[8];
#pragma unroll
    for (int r = 0; r < 8; ++r) { m_st[r] = -1e38f; l_st[r] = 0.0f; }

    const int nkv = q0 + 64;   // causal bound (exclusive)

    if (wv == 0) {
        tdm_load_2d((unsigned)(size_t)(void*)&Ks[0][0][0], Kb, 128, 32, 128, 5, 3);
        tdm_load_2d((unsigned)(size_t)(void*)&Vt[0][0][0], Vb, 32, 128, TKV_, 3, 3);
    }

    for (int j = 0; j < nkv; j += 32) {
        const int p = (j >> 5) & 1;

        if (wv == 0) __builtin_amdgcn_s_wait_tensorcnt(0);  // tile j landed
        __syncthreads();   // publish LDS tile; fences last tile's compute

        const int jn = j + 32;
        if (wv == 0 && jn < nkv) {
            tdm_load_2d((unsigned)(size_t)(void*)&Ks[p ^ 1][0][0],
                        Kb + (size_t)jn * HD_, 128, 32, 128, 5, 3);
            tdm_load_2d((unsigned)(size_t)(void*)&Vt[p ^ 1][0][0],
                        Vb + jn, 32, 128, TKV_, 3, 3);
        }

        // ---- S = Q * K^T over HD=128 (4 chunks of 32) ----
        v8f s0 = {}, s1 = {};
#pragma unroll
        for (int kc = 0; kc < 4; ++kc) {
            v16bf aq  = load_a_frag(&Qs[wm * 16][kc * 32], 136);
            v16bf bk0 = load_b_frag(&Ks[p][0][kc * 32],  1, 136);
            v16bf bk1 = load_b_frag(&Ks[p][16][kc * 32], 1, 136);
            s0 = wmma_bf16(aq, bk0, s0);
            s1 = wmma_bf16(aq, bk1, s1);
        }

        // ---- scale + causal mask (arithmetic; EXEC stays uniform) ----
        float sv0[8], sv1[8];
#pragma unroll
        for (int r = 0; r < 8; ++r) {
            int rowg = q0 + wm * 16 + r + 8 * hi;
            float x0 = s0[r] * scale;
            float x1 = s1[r] * scale;
            if (j + ln      > rowg) x0 = -1e30f;
            if (j + 16 + ln > rowg) x1 = -1e30f;
            sv0[r] = x0; sv1[r] = x1;
        }

        // ---- online softmax: 16-lane row reductions ----
        float mt[8];
#pragma unroll
        for (int r = 0; r < 8; ++r) mt[r] = fmaxf(sv0[r], sv1[r]);
#pragma unroll
        for (int off = 1; off < 16; off <<= 1)
#pragma unroll
            for (int r = 0; r < 8; ++r)
                mt[r] = fmaxf(mt[r], __shfl_xor(mt[r], off, 32));

        float m_new[8], corr[8], p0[8], p1[8], rs[8];
#pragma unroll
        for (int r = 0; r < 8; ++r) {
            m_new[r] = fmaxf(m_st[r], mt[r]);
            corr[r]  = exp2f((m_st[r] - m_new[r]) * L2E);
            p0[r]    = exp2f((sv0[r] - m_new[r]) * L2E);
            p1[r]    = exp2f((sv1[r] - m_new[r]) * L2E);
            rs[r]    = p0[r] + p1[r];
        }
#pragma unroll
        for (int off = 1; off < 16; off <<= 1)
#pragma unroll
            for (int r = 0; r < 8; ++r)
                rs[r] += __shfl_xor(rs[r], off, 32);
#pragma unroll
        for (int r = 0; r < 8; ++r) {
            l_st[r] = l_st[r] * corr[r] + rs[r];
            m_st[r] = m_new[r];
        }
#pragma unroll
        for (int c = 0; c < 4; ++c)
#pragma unroll
            for (int r = 0; r < 8; ++r)
                acc[c][r] *= corr[r];

        // ---- transpose P (D layout -> A layout) via wave-private LDS ----
        __bf16* pw = &Ps[wv][0];
#pragma unroll
        for (int r = 0; r < 8; ++r) {
            int pr = r + 8 * hi;
            pw[pr * 40 + ln]      = (__bf16)p0[r];
            pw[pr * 40 + 16 + ln] = (__bf16)p1[r];
        }
        asm volatile("s_wait_dscnt 0x0" ::: "memory");

        // ---- ctx += P(16x32) * V(32x64 half); Vt is [d][kv] ----
        v16bf pa = load_a_frag(pw, 40);
#pragma unroll
        for (int c = 0; c < 4; ++c) {
            v16bf bv = load_b_frag(&Vt[p][wn * 64 + c * 16][0], 1, 40);
            acc[c] = wmma_bf16(pa, bv, acc[c]);
        }
    }

    // ---- epilogue: normalize and store ctx (B, TQ, HQ, HD) bf16 ----
#pragma unroll
    for (int c = 0; c < 4; ++c) {
#pragma unroll
        for (int r = 0; r < 8; ++r) {
            int t = q0 + wm * 16 + r + 8 * hi;
            int d = wn * 64 + c * 16 + ln;
            float v = acc[c][r] / l_st[r];
            ctx[((size_t)(b * TQ_ + t) * HQ_ + h) * HD_ + d] = (__bf16)v;
        }
    }
}

// ---------------------------------------------------------------------------
// Host-side launch
// ---------------------------------------------------------------------------
extern "C" void kernel_launch(void* const* d_in, const int* in_sizes, int n_in,
                              void* d_out, int out_size, void* d_ws, size_t ws_size,
                              hipStream_t stream) {
    const float* Xq  = (const float*)d_in[0];
    const float* Xkv = (const float*)d_in[1];
    const float* Wq  = (const float*)d_in[2];
    const float* Wk  = (const float*)d_in[3];
    const float* Wv  = (const float*)d_in[4];
    const float* Wo  = (const float*)d_in[5];
    const int* qpos  = (const int*)d_in[6];
    const int* kpos  = (const int*)d_in[7];
    float* out = (float*)d_out;

    char* ws = (char*)d_ws;
    const size_t M = (size_t)B_ * TQ_;               // 4096 rows

    // f32 intermediates
    float* qf = (float*)(ws + 0);                    // 4096x2048
    float* kf = (float*)(ws + 33554432);             // 4096x512
    float* vf = (float*)(ws + 41943040);             // 4096x512
    // bf16 attention operands
    __bf16* Qb = (__bf16*)(ws + 50331648);           // (B,HQ,T,HD)
    __bf16* Kb = (__bf16*)(ws + 67108864);           // (B,HKV,T,HD)
    __bf16* Vb = (__bf16*)(ws + 71303168);           // (B,HKV,HD,T) transposed
    // bf16 GEMM operands (A row-major; weights stored transposed N x K)
    __bf16* Xqb  = (__bf16*)(ws + 75497472);         // 4096x2048
    __bf16* Xkvb = (__bf16*)(ws + 92274688);         // 4096x2048
    __bf16* Wqt  = (__bf16*)(ws + 109051904);        // 2048x2048 (N=2048,K=2048)
    __bf16* Wkt  = (__bf16*)(ws + 117440512);        // 512x2048
    __bf16* Wvt  = (__bf16*)(ws + 119537664);        // 512x2048
    __bf16* Wot  = (__bf16*)(ws + 121634816);        // 2048x2048
    __bf16* ctx  = (__bf16*)(ws + 0);                // reuse qf region

    // 0) one-time conversions (+ weight transposes for NT GEMM)
    cvt_f32_bf16<<<8388608 / 1024, 256, 0, stream>>>(Xq,  Xqb,  8388608);
    cvt_f32_bf16<<<8388608 / 1024, 256, 0, stream>>>(Xkv, Xkvb, 8388608);
    transpose_cvt<<<dim3(64, 64), dim3(32, 8), 0, stream>>>(Wq, Wqt, 2048, 2048);
    transpose_cvt<<<dim3(16, 64), dim3(32, 8), 0, stream>>>(Wk, Wkt, 2048, 512);
    transpose_cvt<<<dim3(16, 64), dim3(32, 8), 0, stream>>>(Wv, Wvt, 2048, 512);
    transpose_cvt<<<dim3(64, 64), dim3(32, 8), 0, stream>>>(Wo, Wot, 2048, 2048);

    // 1) QKV projections: all-bf16 TDM-staged NT WMMA GEMMs
    gemm_bf16<<<dim3(2048 / 64, M / 64), 256, 0, stream>>>(Xqb,  Wqt, qf, (int)M, 2048, 2048);
    gemm_bf16<<<dim3(512  / 64, M / 64), 256, 0, stream>>>(Xkvb, Wkt, kf, (int)M, 512,  2048);
    gemm_bf16<<<dim3(512  / 64, M / 64), 256, 0, stream>>>(Xkvb, Wvt, vf, (int)M, 512,  2048);

    // 2) RoPE (Q,K) + V transpose to (B,HKV,HD,T)
    rope_transpose<<<B_ * TQ_  * HQ_,  64, 0, stream>>>(qf, Qb, qpos, TQ_,  HQ_);
    rope_transpose<<<B_ * TKV_ * HKV_, 64, 0, stream>>>(kf, Kb, kpos, TKV_, HKV_);
    transpose_v<<<dim3(TKV_ / 32, HD_ / 32, B_ * HKV_), dim3(32, 8), 0, stream>>>(vf, Vb);

    // 3) Causal flash attention -> ctx (B,TQ,HQ,HD) bf16
    flash_attn<<<dim3(TQ_ / 64, B_ * HQ_), 256, 0, stream>>>(Qb, Kb, Vb, ctx);

    // 4) Output projection
    gemm_bf16<<<dim3(2048 / 64, M / 64), 256, 0, stream>>>(ctx, Wot, out, (int)M, 2048, 2048);
}
